// CompressionDistortion_18700287606832
// MI455X (gfx1250) — matrix operand order repaired and checked by reading, not compiled, not run on hardware
//
#include <hip/hip_runtime.h>
#include <stdint.h>

typedef __attribute__((ext_vector_type(16))) _Float16 v16h;
typedef __attribute__((ext_vector_type(8)))  float    v8f;

#define NSIG   4096
#define NDET   3840
#define TPB    256
#define NWAVE  8

// db4 scaling filter, its QMF, and both reversed (for inverse transform).
__device__ __constant__ float c_LO[8] = {
    0.23037781330885523f,  0.7148465705525415f,   0.6308807679295904f,
   -0.02798376941698385f, -0.18703481171888114f,  0.030841381835986965f,
    0.032883011666982945f,-0.010597401784997278f };
__device__ __constant__ float c_HI[8] = {
   -0.010597401784997278f,-0.032883011666982945f, 0.030841381835986965f,
    0.18703481171888114f, -0.02798376941698385f, -0.6308807679295904f,
    0.7148465705525415f,  -0.23037781330885523f };
__device__ __constant__ float c_LOR[8] = {
   -0.010597401784997278f, 0.032883011666982945f, 0.030841381835986965f,
   -0.18703481171888114f, -0.02798376941698385f,  0.6308807679295904f,
    0.7148465705525415f,   0.23037781330885523f };
__device__ __constant__ float c_HIR[8] = {
   -0.23037781330885523f,  0.7148465705525415f,  -0.6308807679295904f,
   -0.02798376941698385f,  0.18703481171888114f,  0.030841381835986965f,
   -0.032883011666982945f,-0.010597401784997278f };

// B operand: banded filter matrix. B layout: N = lane&15; h<8 -> K = 8*(lane>>4)+h ; h>=8 -> K = 16 + 8*(lane>>4) + (h-8).
__device__ __forceinline__ v16h build_Bfwd(const float* f, int chunk, int lane) {
    v16h b;
    const int col   = lane & 15;
    const int kbase = (lane >> 4) * 8;
#pragma unroll
    for (int h = 0; h < 16; ++h) {
        int p = chunk * 32 + ((h < 8) ? (kbase + h) : (16 + kbase + (h - 8)));
        int j = p - 2 * col;                 // B[p][n] = f[p-2n]
        float v = (j >= 0 && j < 8) ? f[j] : 0.0f;
        b[h] = (_Float16)v;
    }
    return b;
}
__device__ __forceinline__ v16h build_Binv(const float* f, int lane) {
    v16h b;
    const int col   = lane & 15;
    const int kbase = (lane >> 4) * 8;
#pragma unroll
    for (int h = 0; h < 16; ++h) {
        int p = (h < 8) ? (kbase + h) : (16 + kbase + (h - 8));
        int j = p - col;                     // B[p][n] = f[p-n]
        float v = (j >= 0 && j < 8) ? f[j] : 0.0f;
        b[h] = (_Float16)v;
    }
    return b;
}

// A operand: im2row window. ISA 7.12.2 A layout: M = lane&15;
// h<8 -> K = 8*(lane>>4)+h ; h>=8 -> K = 16 + 8*(lane>>4) + (h-8).
__device__ __forceinline__ v16h build_Afwd(const float* in, int Nin, int base,
                                           int chunk, int lane) {
    v16h a;
    const int t     = lane & 15;
    const int khalf = (lane >> 4) * 8;
    const int mask  = Nin - 1;               // power of two -> periodization
    const int rb    = base + 32 * t + 32 * chunk;
#pragma unroll
    for (int h = 0; h < 16; ++h) {
        int K = (h < 8) ? (khalf + h) : (16 + khalf + (h - 8));
        a[h] = (_Float16)in[(rb + K) & mask];
    }
    return a;
}
// Zero-stuffed (upsampled) source with optional fused soft threshold.
__device__ __forceinline__ v16h build_Ainv(const float* src, int N2, int i0,
                                           int lane, float thr, bool do_soft) {
    v16h a;
    const int t     = lane & 15;
    const int khalf = (lane >> 4) * 8;
    const int mask  = N2 - 1;
    const int rb    = i0 + 16 * t - 7 + N2;  // stay non-negative before mask
#pragma unroll
    for (int h = 0; h < 16; ++h) {
        int K = (h < 8) ? (khalf + h) : (16 + khalf + (h - 8));
        int q = (rb + K) & mask;
        float v = 0.0f;
        if (!(q & 1)) {
            v = src[q >> 1];
            if (do_soft) {
                float av = fabsf(v) - thr;
                av = av > 0.0f ? av : 0.0f;
                v = copysignf(av, v);
            }
        }
        a[h] = (_Float16)v;
    }
    return a;
}

__device__ __forceinline__ void fwd_level(const float* in, float* a_out, float* d_out,
                                          int Nin, int wave, int lane,
                                          v16h Blo0, v16h Blo1, v16h Bhi0, v16h Bhi1) {
    const int ntiles = Nin >> 9;             // (Nin/2)/256 output tiles
    for (int tile = wave; tile < ntiles; tile += NWAVE) {
        const int i0   = tile << 8;
        const int base = i0 << 1;
        v16h a0 = build_Afwd(in, Nin, base, 0, lane);
        v16h a1 = build_Afwd(in, Nin, base, 1, lane);
        v8f clo = {}; v8f chi = {};
        clo = __builtin_amdgcn_wmma_f32_16x16x32_f16(false, a0, false, Blo0, (short)0, clo, false, false);
        clo = __builtin_amdgcn_wmma_f32_16x16x32_f16(false, a1, false, Blo1, (short)0, clo, false, false);
        chi = __builtin_amdgcn_wmma_f32_16x16x32_f16(false, a0, false, Bhi0, (short)0, chi, false, false);
        chi = __builtin_amdgcn_wmma_f32_16x16x32_f16(false, a1, false, Bhi1, (short)0, chi, false, false);
        const int n = lane & 15;
#pragma unroll
        for (int r = 0; r < 8; ++r) {        // D layout: M = r + 8*(lane>=16)
            int oi = i0 + ((r + ((lane >> 4) << 3)) << 4) + n;
            a_out[oi] = clo[r];
            d_out[oi] = chi[r];
        }
    }
}

__device__ __forceinline__ void inv_level(const float* rec, const float* det, float* out,
                                          int n_in, int wave, int lane, float thr,
                                          v16h Blor, v16h Bhir) {
    const int N2     = n_in << 1;
    const int ntiles = N2 >> 8;
    for (int tile = wave; tile < ntiles; tile += NWAVE) {
        const int i0 = tile << 8;
        v16h au = build_Ainv(rec, N2, i0, lane, 0.0f, false);
        v16h av = build_Ainv(det, N2, i0, lane, thr,  true);
        v8f c = {};
        c = __builtin_amdgcn_wmma_f32_16x16x32_f16(false, au, false, Blor, (short)0, c, false, false);
        c = __builtin_amdgcn_wmma_f32_16x16x32_f16(false, av, false, Bhir, (short)0, c, false, false);
        const int n = lane & 15;
#pragma unroll
        for (int r = 0; r < 8; ++r)
            out[i0 + ((r + ((lane >> 4) << 3)) << 4) + n] = c[r];
    }
}

// Exact k-th order statistic of |det[0..n)| via MSB-first 8-bit radix select.
// Bucket localization done by wave 0 with a wave32 shuffle scan (no serial 256-bin walk).
__device__ __forceinline__ float radix_select_abs(const float* det, int n, int k,
                                                  unsigned int* hist, unsigned int* sel,
                                                  int tid) {
    unsigned int prefix = 0;
    int kk = k;
    for (int shift = 24; shift >= 0; shift -= 8) {
        hist[tid] = 0u;                       // TPB == 256 bins
        __syncthreads();
        const unsigned int himask = (shift == 24) ? 0u : (0xFFFFFFFFu << (shift + 8));
        for (int i = tid; i < n; i += TPB) {
            unsigned int key = __float_as_uint(fabsf(det[i]));
            if ((key & himask) == prefix) atomicAdd(&hist[(key >> shift) & 255u], 1u);
        }
        __syncthreads();
        if (tid < 32) {                       // wave 0: 8 bins per lane + shuffle scan
            const unsigned int base = (unsigned)tid * 8u;
            unsigned int cnt[8];
            unsigned int s = 0;
#pragma unroll
            for (int i = 0; i < 8; ++i) { cnt[i] = hist[base + i]; s += cnt[i]; }
            unsigned int inc = s;
#pragma unroll
            for (int off = 1; off < 32; off <<= 1) {
                unsigned int o = __shfl_up(inc, off, 32);
                if (tid >= off) inc += o;
            }
            const unsigned int excl = inc - s;
            const unsigned int kq = (unsigned int)kk;
            if (excl <= kq && kq < inc) {     // exactly one lane hits
                unsigned int cum = excl;
                unsigned int b = base;
#pragma unroll
                for (int i = 0; i < 8; ++i) {
                    if (cum + cnt[i] > kq) { b = base + (unsigned)i; break; }
                    cum += cnt[i];
                }
                sel[0] = b;
                sel[1] = kq - cum;
            }
        }
        __syncthreads();
        prefix |= sel[0] << shift;
        kk = (int)sel[1];
        __syncthreads();
    }
    return __uint_as_float(prefix);
}

__global__ void __launch_bounds__(TPB)
wavelet_compress_kernel(const float* __restrict__ x,
                        const float* __restrict__ dither,
                        float* __restrict__ out) {
    __shared__ float s_x[4096];
    __shared__ float s_det[3840];        // d1[2048] d2[1024] d3[512] d4[256]
    __shared__ float s_a[2048];
    __shared__ float s_b[2048];
    __shared__ unsigned int s_hist[256];
    __shared__ unsigned int s_sel[2];
    __shared__ float s_power;

    const int tid  = threadIdx.x;
    const int lane = tid & 31;
    const int wave = tid >> 5;
    const long long sig = blockIdx.x;
    const float* xg = x      + sig * NSIG;
    const float* dg = dither + sig * NSIG;
    float*       og = out    + sig * NSIG;

    // Warm L2 with the dither stream (consumed only at the very end).
#pragma unroll
    for (int r = 0; r < 4; ++r)
        __builtin_prefetch(dg + tid * 4 + r * 1024, 0, 1);

    // Stage signal into LDS via gfx1250 async global->LDS DMA (ASYNCcnt-tracked):
    // 16B per lane per issue, 4 issues cover the 16 KB signal.
    {
        unsigned int lds_off = (unsigned int)(unsigned long long)(&s_x[tid * 4]);
        const float* gp = xg + tid * 4;
#pragma unroll
        for (int r = 0; r < 4; ++r) {
            asm volatile("global_load_async_to_lds_b128 %0, %1, off"
                         :: "v"(lds_off + (unsigned int)(r * 4096)),
                            "v"(gp + r * 1024)
                         : "memory");
        }
    }

    // Constant banded filter operands (per-lane, built once) — overlaps the DMA.
    v16h Blo0 = build_Bfwd(c_LO, 0, lane);
    v16h Blo1 = build_Bfwd(c_LO, 1, lane);
    v16h Bhi0 = build_Bfwd(c_HI, 0, lane);
    v16h Bhi1 = build_Bfwd(c_HI, 1, lane);
    v16h Blor = build_Binv(c_LOR, lane);
    v16h Bhir = build_Binv(c_HIR, lane);

    asm volatile("s_wait_asynccnt 0x0" ::: "memory");
    __syncthreads();

    // ---- 4-level analysis DWT (WMMA banded GEMM) ----
    fwd_level(s_x, s_a, s_det + 0,    4096, wave, lane, Blo0, Blo1, Bhi0, Bhi1);
    __syncthreads();
    fwd_level(s_a, s_b, s_det + 2048, 2048, wave, lane, Blo0, Blo1, Bhi0, Bhi1);
    __syncthreads();
    fwd_level(s_b, s_a, s_det + 3072, 1024, wave, lane, Blo0, Blo1, Bhi0, Bhi1);
    __syncthreads();
    fwd_level(s_a, s_b, s_det + 3584,  512, wave, lane, Blo0, Blo1, Bhi0, Bhi1);
    __syncthreads();

    // 25th percentile: pos = 0.25*3839 = 959.75 -> lerp order stats 959, 960
    float v0 = radix_select_abs(s_det, NDET, 959, s_hist, s_sel, tid);
    float v1 = radix_select_abs(s_det, NDET, 960, s_hist, s_sel, tid);
    const float thr = 0.25f * v0 + 0.75f * v1;

    // ---- 4-level synthesis with fused soft-threshold ----
    inv_level(s_b, s_det + 3584, s_a,  256, wave, lane, thr, Blor, Bhir);
    __syncthreads();
    inv_level(s_a, s_det + 3072, s_b,  512, wave, lane, thr, Blor, Bhir);
    __syncthreads();
    inv_level(s_b, s_det + 2048, s_a, 1024, wave, lane, thr, Blor, Bhir);
    __syncthreads();
    inv_level(s_a, s_det + 0,    s_x, 2048, wave, lane, thr, Blor, Bhir);
    __syncthreads();

    // ---- quantization at fixed target SNR (30 dB -> snr_lin = 1000) ----
    if (tid == 0) s_power = 0.0f;
    __syncthreads();
    const float4* sx4 = (const float4*)s_x;
    float psum = 0.0f;
#pragma unroll
    for (int r = 0; r < 4; ++r) {
        float4 rv = sx4[tid + r * TPB];
        psum += rv.x * rv.x + rv.y * rv.y + rv.z * rv.z + rv.w * rv.w;
    }
    atomicAdd(&s_power, psum);
    __syncthreads();
    const float power    = s_power * (1.0f / (float)NSIG);
    const float step     = sqrtf(12.0f * power * 1.0e-3f);  // snr_lin = 10^3
    const float inv_step = 1.0f / step;
    const float dscale   = step * 0.1f;
    const float4* dg4 = (const float4*)dg;
    float4*       og4 = (float4*)og;
#pragma unroll
    for (int r = 0; r < 4; ++r) {
        const int i = tid + r * TPB;
        float4 rv = sx4[i];
        float4 dv = dg4[i];
        float4 q;
        q.x = rintf(rv.x * inv_step) * step + (dv.x - 0.5f) * dscale;  // round-half-even
        q.y = rintf(rv.y * inv_step) * step + (dv.y - 0.5f) * dscale;
        q.z = rintf(rv.z * inv_step) * step + (dv.z - 0.5f) * dscale;
        q.w = rintf(rv.w * inv_step) * step + (dv.w - 0.5f) * dscale;
        og4[i] = q;
    }
}

extern "C" void kernel_launch(void* const* d_in, const int* in_sizes, int n_in,
                              void* d_out, int out_size, void* d_ws, size_t ws_size,
                              hipStream_t stream) {
    (void)n_in; (void)out_size; (void)d_ws; (void)ws_size;
    const float* x  = (const float*)d_in[0];
    const float* dn = (const float*)d_in[1];
    float* out      = (float*)d_out;
    const int M = in_sizes[0] / NSIG;   // 64*128 = 8192 signals
    wavelet_compress_kernel<<<dim3(M), dim3(TPB), 0, stream>>>(x, dn, out);
}
